// CAGenerator_50852412784644
// MI455X (gfx1250) — matrix operand order repaired and checked by reading, not compiled, not run on hardware
//
#include <hip/hip_runtime.h>
#include <stdint.h>

// ---------------------------------------------------------------------------
// Neural CA on gfx1250: fused conv3x3(80->512) -> 1x1(512->512,relu) ->
// 1x1(512->80) + residual, per-step alive masking. All GEMMs run on
// v_wmma_f32_16x16x32_bf16 (fp32 accumulate), state kept fp32.
// ~705 GFLOP total, weights L2-resident -> compute bound -> bf16 WMMA path.
// ---------------------------------------------------------------------------

typedef __attribute__((ext_vector_type(16))) __bf16 v16bf;
typedef __attribute__((ext_vector_type(8)))  float  v8f;

#define NB      8
#define CH      80
#define HW      1024           // 32*32
#define HID     512
#define STEPS   64
#define THRESH  0.01f
#define K1      720            // 80 * 9
#define K1C     23             // ceil(736/32), K padded to 736
#define K2C     16             // 512/32
#define ACT_STRIDE 520         // 512 + 8 pad (keeps 16B alignment, dodges banks)

__device__ __forceinline__ unsigned short f2bf(float x) {
    uint32_t u = __float_as_uint(x);
    return (unsigned short)((u + 0x7FFFu + ((u >> 16) & 1u)) >> 16);
}

union Frag16 { uint32_t u[8]; uint4 q[2]; v16bf v; };

__device__ __forceinline__ void loadB32(Frag16& f, const unsigned short* p) {
    const uint4* q = (const uint4*)p;   // 32B-aligned packed fragment
    f.q[0] = q[0];
    f.q[1] = q[1];
}

// ---------------------------------------------------------------------------
// Fused 3-layer step. One block = 16 pixels (half a row of one image).
// ---------------------------------------------------------------------------
__global__ __launch_bounds__(256) void ca_fused_step(
    const float* __restrict__ sIn, float* __restrict__ sOut,
    const unsigned short* __restrict__ pW1,
    const unsigned short* __restrict__ pW2,
    const unsigned short* __restrict__ pW3,
    const float* __restrict__ b1, const float* __restrict__ b2,
    const float* __restrict__ b3)
{
    __shared__ unsigned short raw[3 * 18 * 80];        // [ky][col][c] bf16 halo
    __shared__ unsigned short act[16][ACT_STRIDE];     // inter-layer acts, bf16

    const int tid  = threadIdx.x;
    const int lane = tid & 31;
    const int wave = tid >> 5;
    const int row  = lane & 15;        // A-matrix M row / C-matrix N col
    const int half = lane >> 4;

    const int m   = blockIdx.x;        // 0..511 M-tiles
    const int b   = m >> 6;
    const int rem = m & 63;
    const int y   = rem >> 1;
    const int x0  = (rem & 1) << 4;

    const float* sb = sIn + (size_t)b * CH * HW;

    // ---- gather 3x18x80 halo (tap-major -> channel-contiguous in LDS) ----
    for (int idx = tid; idx < 3 * 18 * 80; idx += 256) {
        int c   = idx / 54;
        int r   = idx - c * 54;
        int ky  = r / 18;
        int col = r - ky * 18;
        int yy  = y + ky - 1;
        int xx  = x0 + col - 1;
        float v = 0.0f;
        if (yy >= 0 && yy < 32 && xx >= 0 && xx < 32)
            v = sb[(c << 10) + (yy << 5) + xx];
        raw[(ky * 18 + col) * 80 + c] = f2bf(v);
    }
    __syncthreads();

    // ================= GEMM1: perception = im2col(state) @ W1 =============
    {
        v8f acc[4] = {};
        for (int kc = 0; kc < K1C; ++kc) {
            Frag16 a;
            // two contiguous 8-channel runs per lane (tap-major K order)
            #pragma unroll
            for (int run = 0; run < 2; ++run) {
                int K0 = kc * 32 + run * 16 + half * 8;
                if (K0 < K1) {
                    int tap = K0 / 80;
                    int c0  = K0 - tap * 80;
                    int ky  = tap / 3;
                    int kx  = tap - ky * 3;
                    a.q[run] = *(const uint4*)&raw[(ky * 18 + row + kx) * 80 + c0];
                } else {
                    a.q[run] = make_uint4(0u, 0u, 0u, 0u);
                }
            }
            #pragma unroll
            for (int nt = 0; nt < 4; ++nt) {
                const unsigned short* bp =
                    pW1 + (((size_t)((wave * 4 + nt) * K1C + kc) * 32 + lane) << 4);
                __builtin_prefetch(bp + 512, 0, 0);   // next K-chunk
                Frag16 bfr;
                loadB32(bfr, bp);
                acc[nt] = __builtin_amdgcn_wmma_f32_16x16x32_bf16(
                    false, a.v, false, bfr.v, (short)0, acc[nt], false, false);
            }
        }
        // write perception + b1 to LDS (bf16)
        #pragma unroll
        for (int nt = 0; nt < 4; ++nt) {
            int n = (wave * 4 + nt) * 16 + row;    // C: N = lane&15
            float bias = b1[n];
            #pragma unroll
            for (int r = 0; r < 8; ++r)
                act[r + half * 8][n] = f2bf(acc[nt][r] + bias);
        }
    }
    __syncthreads();

    // ================= GEMM2: h = relu(perception @ W2 + b2) ==============
    {
        v8f acc[4] = {};
        for (int kc = 0; kc < K2C; ++kc) {
            Frag16 a;
            int K0 = kc * 32 + half * 8;
            a.q[0] = *(const uint4*)&act[row][K0];
            a.q[1] = *(const uint4*)&act[row][K0 + 16];
            #pragma unroll
            for (int nt = 0; nt < 4; ++nt) {
                const unsigned short* bp =
                    pW2 + (((size_t)((wave * 4 + nt) * K2C + kc) * 32 + lane) << 4);
                __builtin_prefetch(bp + 512, 0, 0);
                Frag16 bfr;
                loadB32(bfr, bp);
                acc[nt] = __builtin_amdgcn_wmma_f32_16x16x32_bf16(
                    false, a.v, false, bfr.v, (short)0, acc[nt], false, false);
            }
        }
        __syncthreads();   // all waves done reading act before overwrite
        #pragma unroll
        for (int nt = 0; nt < 4; ++nt) {
            int n = (wave * 4 + nt) * 16 + row;
            float bias = b2[n];
            #pragma unroll
            for (int r = 0; r < 8; ++r) {
                float v = acc[nt][r] + bias;
                act[r + half * 8][n] = f2bf(v > 0.0f ? v : 0.0f);
            }
        }
    }
    __syncthreads();

    // ============ GEMM3: updates = h @ W3 + b3; state += updates ==========
    if (wave < 5) {
        v8f acc = {};
        for (int kc = 0; kc < K2C; ++kc) {
            Frag16 a;
            int K0 = kc * 32 + half * 8;
            a.q[0] = *(const uint4*)&act[row][K0];
            a.q[1] = *(const uint4*)&act[row][K0 + 16];
            const unsigned short* bp =
                pW3 + (((size_t)(wave * K2C + kc) * 32 + lane) << 4);
            Frag16 bfr;
            loadB32(bfr, bp);
            acc = __builtin_amdgcn_wmma_f32_16x16x32_bf16(
                false, a.v, false, bfr.v, (short)0, acc, false, false);
        }
        int ch = wave * 16 + row;                 // C: N = channel
        float bias = b3[ch];
        const float* ib = sIn  + (size_t)b * CH * HW + (ch << 10) + (y << 5);
        float*       ob = sOut + (size_t)b * CH * HW + (ch << 10) + (y << 5);
        #pragma unroll
        for (int r = 0; r < 8; ++r) {
            int px = x0 + r + half * 8;           // C: M = pixel in tile
            ob[px] = ib[px] + acc[r] + bias;
        }
    }
}

// ---------------------------------------------------------------------------
// Alive mask: living = (maxpool3(pre ch0)>T) & (maxpool3(post ch0)>T)
// ---------------------------------------------------------------------------
__global__ __launch_bounds__(256) void ca_mask(const float* __restrict__ sIn,
                                               float* __restrict__ sOut)
{
    __shared__ float pin[HW];
    __shared__ float pout[HW];
    const int b = blockIdx.x;
    const float* ib = sIn  + (size_t)b * CH * HW;
    float*       ob = sOut + (size_t)b * CH * HW;

    for (int i = threadIdx.x; i < HW; i += 256) { pin[i] = ib[i]; pout[i] = ob[i]; }
    __syncthreads();

    float liv[4];
    #pragma unroll
    for (int j = 0; j < 4; ++j) {
        int p  = threadIdx.x + j * 256;
        int py = p >> 5, px = p & 31;
        float mpre = -1e30f, mpost = -1e30f;
        #pragma unroll
        for (int dy = -1; dy <= 1; ++dy)
            #pragma unroll
            for (int dx = -1; dx <= 1; ++dx) {
                int yy = py + dy, xx = px + dx;
                if (yy >= 0 && yy < 32 && xx >= 0 && xx < 32) {
                    mpre  = fmaxf(mpre,  pin[(yy << 5) + xx]);
                    mpost = fmaxf(mpost, pout[(yy << 5) + xx]);
                }
            }
        liv[j] = (mpre > THRESH && mpost > THRESH) ? 1.0f : 0.0f;
    }
    for (int c = 0; c < CH; ++c)
        #pragma unroll
        for (int j = 0; j < 4; ++j) {
            int p = threadIdx.x + j * 256;
            ob[(c << 10) + p] *= liv[j];
        }
}

// ---------------------------------------------------------------------------
// Init: zeros; seed center pixel (ch0=1, ch1..79 = z[:, :79])
// ---------------------------------------------------------------------------
__global__ void ca_init(const float* __restrict__ z, float* __restrict__ s)
{
    int i = blockIdx.x * 256 + threadIdx.x;
    if (i >= NB * CH * HW) return;
    int b = i / (CH * HW);
    int r = i - b * (CH * HW);
    int c = r >> 10;
    int p = r & 1023;
    float v = 0.0f;
    if (p == (16 * 32 + 16)) v = (c == 0) ? 1.0f : z[b * 100 + (c - 1)];
    s[i] = v;
}

__global__ void ca_out(const float* __restrict__ s, float* __restrict__ out)
{
    int i = blockIdx.x * 256 + threadIdx.x;
    if (i >= NB * HW) return;
    int b = i >> 10;
    out[i] = s[(size_t)b * CH * HW + (i & 1023)];
}

// ---------------------------------------------------------------------------
// Weight pre-packing into WMMA B-fragment order (bf16).
// B 32x16 layout: lane l holds col (l&15); K = (e>>3)*16 + (l>>4)*8 + (e&7) within chunk.
// ---------------------------------------------------------------------------
__global__ void pack_w1(const float* __restrict__ W1, unsigned short* __restrict__ dst)
{
    int i = blockIdx.x * 256 + threadIdx.x;
    if (i >= 32 * K1C * 512) return;
    int e    = i & 15;
    int lane = (i >> 4) & 31;
    int t    = i >> 9;
    int kc   = t % K1C;
    int nt   = t / K1C;
    int n    = nt * 16 + (lane & 15);
    int K    = kc * 32 + (e >> 3) * 16 + (lane >> 4) * 8 + (e & 7);     // tap-major: K = tap*80 + c
    float v  = 0.0f;
    if (K < K1) {
        int tap = K / 80;
        int c   = K - tap * 80;
        int ky  = tap / 3;
        int kx  = tap - ky * 3;
        v = W1[((n * 80 + c) * 3 + ky) * 3 + kx];
    }
    dst[i] = f2bf(v);
}

__global__ void pack_lin(const float* __restrict__ W, unsigned short* __restrict__ dst,
                         int ntiles, int kdim)
{
    int kcc = kdim >> 5;
    int total = ntiles * kcc * 512;
    int i = blockIdx.x * 256 + threadIdx.x;
    if (i >= total) return;
    int e    = i & 15;
    int lane = (i >> 4) & 31;
    int t    = i >> 9;
    int kc   = t % kcc;
    int nt   = t / kcc;
    int n    = nt * 16 + (lane & 15);
    int K    = kc * 32 + (e >> 3) * 16 + (lane >> 4) * 8 + (e & 7);
    dst[i] = f2bf(W[n * kdim + K]);
}

// ---------------------------------------------------------------------------
extern "C" void kernel_launch(void* const* d_in, const int* in_sizes, int n_in,
                              void* d_out, int out_size, void* d_ws, size_t ws_size,
                              hipStream_t stream)
{
    const float* z  = (const float*)d_in[0];
    const float* W1 = (const float*)d_in[1];
    const float* b1 = (const float*)d_in[2];
    const float* W2 = (const float*)d_in[3];
    const float* b2 = (const float*)d_in[4];
    const float* W3 = (const float*)d_in[5];
    const float* b3 = (const float*)d_in[6];

    (void)in_sizes; (void)n_in; (void)out_size; (void)ws_size;

    char* ws = (char*)d_ws;
    const size_t stateBytes = (size_t)NB * CH * HW * sizeof(float);
    float* sA = (float*)ws;  ws += stateBytes;
    float* sB = (float*)ws;  ws += stateBytes;
    unsigned short* pW1 = (unsigned short*)ws;  ws += (size_t)32 * K1C * 512 * 2;
    unsigned short* pW2 = (unsigned short*)ws;  ws += (size_t)32 * K2C * 512 * 2;
    unsigned short* pW3 = (unsigned short*)ws;  ws += (size_t)5  * K2C * 512 * 2;

    pack_w1 <<<(32 * K1C * 512 + 255) / 256, 256, 0, stream>>>(W1, pW1);
    pack_lin<<<(32 * K2C * 512 + 255) / 256, 256, 0, stream>>>(W2, pW2, 32, 512);
    pack_lin<<<(5  * K2C * 512 + 255) / 256, 256, 0, stream>>>(W3, pW3, 5, 512);
    ca_init <<<(NB * CH * HW + 255) / 256, 256, 0, stream>>>(z, sA);

    float* cur = sA;
    float* nxt = sB;
    for (int s = 0; s < STEPS; ++s) {
        ca_fused_step<<<512, 256, 0, stream>>>(cur, nxt, pW1, pW2, pW3, b1, b2, b3);
        ca_mask<<<NB, 256, 0, stream>>>(cur, nxt);
        float* t = cur; cur = nxt; nxt = t;
    }
    ca_out<<<(NB * HW + 255) / 256, 256, 0, stream>>>(cur, (float*)d_out);
}
